// GRUExtractor_10840497455120
// MI455X (gfx1250) — matrix operand order repaired and checked by reading, not compiled, not run on hardware
//
#include <hip/hip_runtime.h>
#include <math.h>

typedef __attribute__((ext_vector_type(16))) __bf16 v16bf;
typedef __attribute__((ext_vector_type(8)))  __bf16 v8bf;
typedef __attribute__((ext_vector_type(8)))  float  v8f;

#define Bn   256
#define Sn   512
#define DIN  128
#define Hn   512

// ---------------- bf16 helpers (hardware cvt via fptrunc/fpext) ----------------
static __device__ __forceinline__ float bf2f(__bf16 x) { return (float)x; }
static __device__ __forceinline__ __bf16 f2bf(float f) { return (__bf16)f; }

// Branch-free transcendentals on native v_exp_f32 (2^x) + v_rcp_f32.
// sigmoid: 3 VALU; tanh: 4 VALU. Saturate correctly for |x| large.
#define LOG2E  1.44269504088896340736f
static __device__ __forceinline__ float sigmoid_(float x) {
  return __builtin_amdgcn_rcpf(1.0f + __builtin_amdgcn_exp2f(-LOG2E * x));
}
static __device__ __forceinline__ float tanh_(float x) {
  return 1.0f - 2.0f * __builtin_amdgcn_rcpf(1.0f + __builtin_amdgcn_exp2f(2.0f * LOG2E * x));
}

// A fragment (16x32 bf16). `row` = base + (lane&15)*stride + 16*(lane>>4).
// Element e -> K=kbase+16*hi+e (contiguous, matches B). Two 16B loads.
static __device__ __forceinline__ v16bf load_a_bf(const __bf16* row, int kbase) {
  v8bf lo  = *(const v8bf*)(row + kbase);
  v8bf hi8 = *(const v8bf*)(row + kbase + 8);
  return __builtin_shufflevector(lo, hi8, 0, 1, 2, 3, 4, 5, 6, 7,
                                 8, 9, 10, 11, 12, 13, 14, 15);
}
// B fragment (32x16 bf16) from row-major W[N_out, K]: B[k][n] = W[n_glob][k].
// Lane n = lane&15, element e -> K = kbase + 16*hi + e : 16 contiguous bf16.
static __device__ __forceinline__ v16bf load_b_bf(const __bf16* base, int row, int rs,
                                                  int kbase, int hi) {
  return *(const v16bf*)(base + (long)row * rs + kbase + 16 * hi);
}
static __device__ __forceinline__ v8f wmma_bf16(v16bf a, v16bf b, v8f c) {
  return __builtin_amdgcn_wmma_f32_16x16x32_bf16(false, a, false, b, (short)0, c,
                                                 false, false);
}

// ---------------- K0: fp32 -> bf16 weight conversion ----------------
__global__ __launch_bounds__(256) void cvt_kernel(const float* __restrict__ src,
                                                  __bf16* __restrict__ dst, int n) {
  int i = blockIdx.x * 256 + threadIdx.x;
  if (i < n) dst[i] = f2bf(src[i]);
}

// ---------------- K1: X = tanh(input @ Wpre^T + bpre)  [131072,128]x[128,512] ----
__global__ __launch_bounds__(256) void pre_kernel(const float* __restrict__ input,
    const __bf16* __restrict__ Wpre, const float* __restrict__ bpre,
    __bf16* __restrict__ X) {
  const int wave = threadIdx.x >> 5, lane = threadIdx.x & 31;
  const int n = lane & 15, hi = lane >> 4;
  const int idx = blockIdx.x * 8 + wave;           // one 16x16 tile per wave
  const int m0 = (idx >> 5) * 16;                  // 32 N-tiles
  const int n0 = (idx & 31) * 16;
  const float* arow = input + (long)(m0 + n) * DIN + 16 * hi;
  v8f acc = {};
  #pragma unroll
  for (int kb = 0; kb < DIN / 32; ++kb) {
    const int kbase = kb * 32;
    v16bf a;
    #pragma unroll
    for (int e = 0; e < 8; ++e) a[e] = f2bf(arow[kbase + e]);
    #pragma unroll
    for (int e = 0; e < 8; ++e) a[8 + e] = f2bf(arow[kbase + 8 + e]);
    acc = wmma_bf16(a, load_b_bf(Wpre, n0 + n, DIN, kbase, hi), acc);
  }
  const float bias = bpre[n0 + n];
  __bf16* xp = X + (long)(m0 + 8 * hi) * Hn + n0 + n;
  #pragma unroll
  for (int v = 0; v < 8; ++v) xp[v * Hn] = f2bf(tanh_(acc[v] + bias));
}

// ---------------- K2: gi = X @ Wih^T + bih  [131072,512]x[512,1536] -------------
__global__ __launch_bounds__(256) void gi_kernel(const __bf16* __restrict__ X,
    const __bf16* __restrict__ Wih, const float* __restrict__ bih,
    __bf16* __restrict__ gi) {
  const int wave = threadIdx.x >> 5, lane = threadIdx.x & 31;
  const int n = lane & 15, hi = lane >> 4;
  const int idx = blockIdx.x * 8 + wave;
  const int m0 = (idx / 96) * 16;                  // 96 N-tiles
  const int n0 = (idx % 96) * 16;
  const __bf16* arow = X + (long)(m0 + n) * Hn + 16 * hi;
  const __bf16* brow = Wih + (long)(n0 + n) * Hn + 16 * hi;
  v8f acc = {};
  #pragma unroll 4
  for (int kb = 0; kb < Hn / 32; ++kb) {
    const int kbase = kb * 32;
    acc = wmma_bf16(load_a_bf(arow, kbase), *(const v16bf*)(brow + kbase), acc);
  }
  const float bias = bih[n0 + n];
  __bf16* gp = gi + (long)(m0 + 8 * hi) * (3 * Hn) + n0 + n;
  #pragma unroll
  for (int v = 0; v < 8; ++v) gp[v * (3 * Hn)] = f2bf(acc[v] + bias);
}

// ---------------- K3: persistent GRU recurrence + after-GEMM --------------------
// 16 blocks x 256 threads; block owns 16 batch rows; wave owns 64 H-columns and
// computes its r/z/n gate tiles so the GRU update fuses in registers. h_old for a
// lane's cells is the value that same lane produced last step -> kept in VGPRs.
__global__ __launch_bounds__(256) void gru_kernel(const __bf16* __restrict__ gi,
    const float* __restrict__ bhh, const __bf16* __restrict__ Whh,
    const __bf16* __restrict__ Waft, const float* __restrict__ baft,
    float* __restrict__ out) {
  __shared__ __bf16 hb[16 * Hn];   // bf16 copy of h for the WMMA A path (16 KB)
  const int tid = threadIdx.x;
  for (int i = tid; i < 16 * Hn; i += 256) hb[i] = f2bf(0.0f);
  __syncthreads();

  const int wave = tid >> 5, lane = tid & 31;
  const int n = lane & 15, hi = lane >> 4;
  const int b0 = blockIdx.x * 16;
  const int c0 = wave * 64 + n;                    // lane's column (plus j*16)
  const v8f vzero = {0.f, 0.f, 0.f, 0.f, 0.f, 0.f, 0.f, 0.f};

  float br[4], bz[4], bn[4];
  #pragma unroll
  for (int j = 0; j < 4; ++j) {
    const int c = c0 + j * 16;
    br[j] = bhh[c]; bz[j] = bhh[Hn + c]; bn[j] = bhh[2 * Hn + c];
  }
  // W_hh base pointers for the r-gate; z/n gates are constant +Hn*Hn offsets
  // (524288 B / 1048576 B -> inside the 24-bit signed immediate range).
  const __bf16* wr[4];
  #pragma unroll
  for (int j = 0; j < 4; ++j) wr[j] = Whh + (long)(c0 + j * 16) * Hn + 16 * hi;
  // gi row pointer per accumulator row v; advanced by 3H each timestep.
  const __bf16* gp[8];
  #pragma unroll
  for (int v = 0; v < 8; ++v)
    gp[v] = gi + (long)(b0 + v + 8 * hi) * Sn * (3 * Hn) + c0;

  const __bf16* arow = hb + n * Hn + 16 * hi;      // shared A row for this lane
  __bf16* hwp = hb + (8 * hi) * Hn + c0;           // this lane's h write base

  float hprev[4][8];
  #pragma unroll
  for (int j = 0; j < 4; ++j)
    #pragma unroll
    for (int v = 0; v < 8; ++v) hprev[j][v] = 0.0f;

  for (int t = 0; t < Sn; ++t) {
    v8f ar[4], az[4], an_[4];
    #pragma unroll
    for (int j = 0; j < 4; ++j) { ar[j] = vzero; az[j] = vzero; an_[j] = vzero; }
    #pragma unroll 4
    for (int kb = 0; kb < Hn / 32; ++kb) {
      const int kbase = kb * 32;
      const v16bf a = load_a_bf(arow, kbase);
      #pragma unroll
      for (int j = 0; j < 4; ++j) {
        ar[j]  = wmma_bf16(a, *(const v16bf*)(wr[j] + kbase),                ar[j]);
        az[j]  = wmma_bf16(a, *(const v16bf*)(wr[j] + Hn * Hn + kbase),     az[j]);
        an_[j] = wmma_bf16(a, *(const v16bf*)(wr[j] + 2 * Hn * Hn + kbase), an_[j]);
      }
    }
    #pragma unroll
    for (int j = 0; j < 4; ++j) {
      #pragma unroll
      for (int v = 0; v < 8; ++v) {
        const float gir = bf2f(gp[v][j * 16]);
        const float giz = bf2f(gp[v][Hn + j * 16]);
        const float gin = bf2f(gp[v][2 * Hn + j * 16]);
        const float r = sigmoid_(gir + ar[j][v] + br[j]);
        const float z = sigmoid_(giz + az[j][v] + bz[j]);
        const float nn = tanh_(gin + r * (an_[j][v] + bn[j]));
        hprev[j][v] = (1.0f - z) * nn + z * hprev[j][v];
      }
    }
    __syncthreads();                               // all reads of h done
    #pragma unroll
    for (int j = 0; j < 4; ++j)
      #pragma unroll
      for (int v = 0; v < 8; ++v)
        hwp[v * Hn + j * 16] = f2bf(hprev[j][v]);
    __syncthreads();                               // h ready for next step
    #pragma unroll
    for (int v = 0; v < 8; ++v) gp[v] += 3 * Hn;
  }

  // y = tanh(tanh(h_last) @ Wafter^T + bafter)
  #pragma unroll
  for (int j = 0; j < 4; ++j)
    #pragma unroll
    for (int v = 0; v < 8; ++v)
      hwp[v * Hn + j * 16] = f2bf(tanh_(hprev[j][v]));
  __syncthreads();

  v8f acc[4];
  #pragma unroll
  for (int j = 0; j < 4; ++j) acc[j] = vzero;
  const __bf16* wa[4];
  #pragma unroll
  for (int j = 0; j < 4; ++j) wa[j] = Waft + (long)(c0 + j * 16) * Hn + 16 * hi;
  #pragma unroll 4
  for (int kb = 0; kb < Hn / 32; ++kb) {
    const int kbase = kb * 32;
    const v16bf a = load_a_bf(arow, kbase);
    #pragma unroll
    for (int j = 0; j < 4; ++j)
      acc[j] = wmma_bf16(a, *(const v16bf*)(wa[j] + kbase), acc[j]);
  }
  float* op = out + (long)(b0 + 8 * hi) * Hn + c0;
  #pragma unroll
  for (int j = 0; j < 4; ++j) {
    const float bias = baft[c0 + j * 16];
    #pragma unroll
    for (int v = 0; v < 8; ++v)
      op[v * Hn + j * 16] = tanh_(acc[j][v] + bias);
  }
}

// ---------------- launch ----------------
extern "C" void kernel_launch(void* const* d_in, const int* in_sizes, int n_in,
                              void* d_out, int out_size, void* d_ws, size_t ws_size,
                              hipStream_t stream) {
  (void)in_sizes; (void)n_in; (void)out_size; (void)ws_size;
  const float* input  = (const float*)d_in[0];
  const float* Wpre   = (const float*)d_in[1];
  const float* bpre   = (const float*)d_in[2];
  const float* Wih    = (const float*)d_in[3];
  const float* bih    = (const float*)d_in[4];
  const float* Whh    = (const float*)d_in[5];
  const float* bhh    = (const float*)d_in[6];
  const float* Wafter = (const float*)d_in[7];
  const float* bafter = (const float*)d_in[8];

  char* ws = (char*)d_ws;
  __bf16* wpre_b = (__bf16*)(ws + 0);           //  65536 bf16
  __bf16* wih_b  = (__bf16*)(ws + 131072);      // 786432 bf16
  __bf16* whh_b  = (__bf16*)(ws + 1703936);     // 786432 bf16
  __bf16* waft_b = (__bf16*)(ws + 3276800);     // 262144 bf16
  __bf16* Xb     = (__bf16*)(ws + 3801088);     // 131072*512  bf16 (128 MB)
  __bf16* gib    = (__bf16*)(ws + 138018816);   // 131072*1536 bf16 (384 MB)

  cvt_kernel<<<(65536 + 255) / 256, 256, 0, stream>>>(Wpre, wpre_b, 65536);
  cvt_kernel<<<(786432 + 255) / 256, 256, 0, stream>>>(Wih, wih_b, 786432);
  cvt_kernel<<<(786432 + 255) / 256, 256, 0, stream>>>(Whh, whh_b, 786432);
  cvt_kernel<<<(262144 + 255) / 256, 256, 0, stream>>>(Wafter, waft_b, 262144);

  // pre: (131072/16)*(512/16) tiles / 8 waves = 32768 blocks
  pre_kernel<<<32768, 256, 0, stream>>>(input, wpre_b, bpre, Xb);
  // gi: (131072/16)*(1536/16) tiles / 8 waves = 98304 blocks
  gi_kernel<<<98304, 256, 0, stream>>>(Xb, wih_b, bih, gib);
  // recurrence + after: 16 persistent blocks (one per 16-row batch tile)
  gru_kernel<<<16, 256, 0, stream>>>(gib, bhh, whh_b, waft_b, bafter, (float*)d_out);
}
